// Experts_2405181686579
// MI455X (gfx1250) — matrix operand order repaired and checked by reading, not compiled, not run on hardware
//
#include <hip/hip_runtime.h>

// Problem constants (match reference)
#define BB   16
#define EE   8
#define DD   512
#define FF   2048
#define CC   1024
#define ROWS 64      // row tile per workgroup
#define FC   64      // F chunk width
#define XS   520     // LDS row stride for Xs / W1t (512 + 8 pad, keeps 16B alignment)
#define WS2  72      // LDS row stride for W2t / Hs (64 + 8 pad)

typedef __attribute__((ext_vector_type(16))) __bf16 v16bf;
typedef __attribute__((ext_vector_type(8)))  float  v8f;
typedef __attribute__((ext_vector_type(4)))  float  f32x4;
typedef __attribute__((ext_vector_type(2)))  unsigned int u32x2;
typedef __attribute__((ext_vector_type(4)))  int    i32x4v;

typedef __attribute__((address_space(1))) i32x4v* gp128_t;  // global int4*
typedef __attribute__((address_space(3))) i32x4v* lp128_t;  // LDS int4*

#if defined(__gfx1250__) && __has_builtin(__builtin_amdgcn_global_load_async_to_lds_b128)
#define ASYNC_LDS 1
#else
#define ASYNC_LDS 0
#endif

union Frag { v16bf v; uint4 q[2]; };

__device__ __forceinline__ unsigned short f2bf(float f) {
  union { float f; unsigned int u; } c; c.f = f;
  unsigned int u = c.u;
  unsigned int r = (u + 0x7FFFu + ((u >> 16) & 1u)) >> 16;  // RNE
  return (unsigned short)r;
}

__device__ __forceinline__ unsigned int pk_bf16(float a, float b) {
  return ((unsigned int)f2bf(b) << 16) | f2bf(a);
}

__device__ __forceinline__ v8f wmma_bf16(v16bf a, v16bf b, v8f c) {
  return __builtin_amdgcn_wmma_f32_16x16x32_bf16(false, a, false, b, (short)0, c,
                                                 false, false);
}

__device__ __forceinline__ float tanh_fast(float z) {
  z = fminf(10.0f, fmaxf(-10.0f, z));
#if __has_builtin(__builtin_amdgcn_tanh_f32)
  return __builtin_amdgcn_tanh_f32(z);
#else
  float e = __builtin_amdgcn_exp2f(z * -2.8853900817779268f);
  return (1.0f - e) * __builtin_amdgcn_rcpf(1.0f + e);
#endif
}

__device__ __forceinline__ float gelu_tanh(float x) {
  float x3 = x * x * x;
  return 0.5f * x * (1.0f + tanh_fast(0.7978845608028654f * (x + 0.044715f * x3)));
}

// 16-byte global->LDS copy: async DMA (ASYNCcnt) on CDNA5, sync fallback otherwise.
__device__ __forceinline__ void copy16_g2l(const unsigned short* gsrc, unsigned short* ldst) {
#if ASYNC_LDS
  __builtin_amdgcn_global_load_async_to_lds_b128(
      (gp128_t)(unsigned long long)gsrc,
      (lp128_t)(unsigned int)(unsigned long long)ldst,  // flat LDS ptr -> 32-bit LDS offset
      0, 0);
#else
  *(uint4*)ldst = *(const uint4*)gsrc;
#endif
}

__device__ __forceinline__ void wait_async_then_barrier() {
#if ASYNC_LDS
  asm volatile("s_wait_asynccnt 0" ::: "memory");
#endif
  __syncthreads();
}

// ---- pre-pass: transpose + f32->bf16 convert, (E,R,C) f32 -> (E,C,R) bf16 ----
__global__ __launch_bounds__(256)
void transpose_cvt_kernel(const float* __restrict__ src, unsigned short* __restrict__ dst,
                          int R, int C) {
  __shared__ float tile[32][33];
  const int e  = blockIdx.z;
  const int c0 = blockIdx.x * 32;
  const int r0 = blockIdx.y * 32;
  const size_t base = (size_t)e * R * C;
  #pragma unroll
  for (int i = 0; i < 4; ++i) {
    const int r = r0 + threadIdx.y + 8 * i;
    tile[threadIdx.y + 8 * i][threadIdx.x] =
        __builtin_nontemporal_load(src + base + (size_t)r * C + c0 + threadIdx.x);
  }
  __syncthreads();
  #pragma unroll
  for (int i = 0; i < 4; ++i) {
    const int c = c0 + threadIdx.y + 8 * i;
    // regular (RT) store: bf16 weights should stay resident in L2 for the main kernel
    dst[base + (size_t)c * R + r0 + threadIdx.x] = f2bf(tile[threadIdx.x][threadIdx.y + 8 * i]);
  }
}

__global__ __launch_bounds__(256, 1)
void moe_expert_ffn_kernel(const float* __restrict__ xin,
                           const unsigned short* __restrict__ w1t,  // (E,F,D) bf16
                           const float* __restrict__ b1p,
                           const unsigned short* __restrict__ w2t,  // (E,D,F) bf16
                           const float* __restrict__ b2p,
                           float* __restrict__ out) {
  __shared__ __align__(16) unsigned short Xs[ROWS * XS];   // X tile, bf16 [row][k]
  __shared__ __align__(16) unsigned short Ws[DD * WS2];    // union: W1t [f][d]@XS / W2t [d][f]@WS2
  __shared__ __align__(16) unsigned short Hs[ROWS * WS2];  // GeLU(H) chunk, bf16 [row][f_local]

  const int t    = threadIdx.x;
  const int lane = t & 31;
  const int wv   = t >> 5;
  const int lo   = lane & 15;
  const int hi   = lane >> 4;

  const int wg = blockIdx.x;
  const int rt = wg & 15;
  const int e  = (wg >> 4) & 7;
  const int b  = wg >> 7;

  const size_t row0 = ((size_t)(b * EE + e)) * CC + (size_t)rt * ROWS;

  // ---- stage X tile (64 x 512 f32 -> bf16 LDS), non-temporal (streamed once) ----
  {
    const int cq   = (t & 127) << 2;
    const int rsub = t >> 7;
    for (int p = 0; p < 32; ++p) {
      const int r = p * 2 + rsub;
      const f32x4 v =
          __builtin_nontemporal_load((const f32x4*)(xin + (row0 + r) * DD + cq));
      u32x2 pk;
      pk.x = pk_bf16(v.x, v.y);
      pk.y = pk_bf16(v.z, v.w);
      *(u32x2*)&Xs[r * XS + cq] = pk;
    }
  }

  const int tr     = wv >> 1;          // output tile-row (GEMM1 & GEMM2)
  const int tcb    = (wv & 1) * 2;     // GEMM1: 2 tile-cols per wave
  const int tcbase = (wv & 1) * 16;    // GEMM2: 16 tile-cols per wave

  v8f yacc[16];
  #pragma unroll
  for (int j = 0; j < 16; ++j) yacc[j] = (v8f){};

  for (int fc = 0; fc < FF; fc += FC) {
    __syncthreads();  // prev chunk done reading Ws/Hs

    // ---- stage W1t chunk: 64 rows (f) x 512 (d) bf16, async global->LDS ----
    for (int p = 0; p < 16; ++p) {
      const int id  = p * 256 + t;     // 4096 16B chunks
      const int row = id >> 6;         // f_local 0..63
      const int cb  = id & 63;         // 16B block in row
      copy16_g2l(w1t + ((size_t)e * FF + fc + row) * DD + cb * 8,
                 &Ws[row * XS + cb * 8]);
    }
    wait_async_then_barrier();

    // ---- GEMM1: Hc(64x64) = X(64x512) @ W1c(512x64), bf16 WMMA f32 acc ----
    v8f hacc[2];
    hacc[0] = (v8f){};
    hacc[1] = (v8f){};
    const int arow = (tr * 16 + lo) * XS;
    #pragma unroll 4
    for (int kk = 0; kk < DD; kk += 32) {
      Frag a, bf0, bf1;
      const int akb = kk + (hi << 4);
      a.q[0] = *(const uint4*)&Xs[arow + akb];
      a.q[1] = *(const uint4*)&Xs[arow + akb + 8];
      const int bkb = kk + (hi << 4);
      const int br0 = (tcb * 16 + lo) * XS + bkb;
      const int br1 = ((tcb + 1) * 16 + lo) * XS + bkb;
      bf0.q[0] = *(const uint4*)&Ws[br0];
      bf0.q[1] = *(const uint4*)&Ws[br0 + 8];
      bf1.q[0] = *(const uint4*)&Ws[br1];
      bf1.q[1] = *(const uint4*)&Ws[br1 + 8];
      hacc[0] = wmma_bf16(a.v, bf0.v, hacc[0]);
      hacc[1] = wmma_bf16(a.v, bf1.v, hacc[1]);
    }

    // ---- fused bias + tanh-GeLU -> Hs (bf16) ----
    #pragma unroll
    for (int tt = 0; tt < 2; ++tt) {
      const int nloc = (tcb + tt) * 16 + lo;
      const float bias = b1p[(size_t)e * FF + fc + nloc];
      #pragma unroll
      for (int vv = 0; vv < 8; vv += 2) {
        const int m = tr * 16 + vv + hi * 8;       // rows m, m+1
        const float g0 = gelu_tanh(hacc[tt][vv] + bias);
        const float g1 = gelu_tanh(hacc[tt][vv + 1] + bias);
        const unsigned int u = pk_bf16(g0, g1);
        Hs[m * WS2 + nloc]       = (unsigned short)u;
        Hs[(m + 1) * WS2 + nloc] = (unsigned short)(u >> 16);
      }
    }
    __syncthreads();  // Hs ready; everyone done reading W1t

    // ---- stage W2t chunk: 512 rows (d) x 64 (f) bf16, async global->LDS ----
    for (int p = 0; p < 16; ++p) {
      const int id  = p * 256 + t;     // 4096 16B chunks
      const int row = id >> 3;         // d 0..511
      const int cb  = id & 7;          // 16B block in row
      copy16_g2l(w2t + ((size_t)e * DD + row) * FF + fc + cb * 8,
                 &Ws[row * WS2 + cb * 8]);
    }
    wait_async_then_barrier();

    // ---- GEMM2: Y(64x512) += Hc(64x64) @ W2c(64x512) ----
    const int hrow = (tr * 16 + lo) * WS2;
    #pragma unroll
    for (int kk = 0; kk < FC; kk += 32) {
      Frag a;
      const int akb = kk + (hi << 4);
      a.q[0] = *(const uint4*)&Hs[hrow + akb];
      a.q[1] = *(const uint4*)&Hs[hrow + akb + 8];
      const int bkb = kk + (hi << 4);
      #pragma unroll
      for (int j = 0; j < 16; ++j) {
        Frag bb;
        const int nrow = ((tcbase + j) * 16 + lo) * WS2 + bkb;
        bb.q[0] = *(const uint4*)&Ws[nrow];
        bb.q[1] = *(const uint4*)&Ws[nrow + 8];
        yacc[j] = wmma_bf16(a.v, bb.v, yacc[j]);
      }
    }
  }

  // ---- epilogue: + b2, non-temporal store f32 (streamed once) ----
  #pragma unroll
  for (int j = 0; j < 16; ++j) {
    const int n = (tcbase + j) * 16 + lo;
    const float bias = b2p[(size_t)e * DD + n];
    #pragma unroll
    for (int vv = 0; vv < 8; ++vv) {
      const int m = tr * 16 + vv + hi * 8;
      __builtin_nontemporal_store(yacc[j][vv] + bias, out + (row0 + m) * DD + n);
    }
  }
}

extern "C" void kernel_launch(void* const* d_in, const int* in_sizes, int n_in,
                              void* d_out, int out_size, void* d_ws, size_t ws_size,
                              hipStream_t stream) {
  (void)in_sizes; (void)n_in; (void)out_size; (void)ws_size;
  const float* xin = (const float*)d_in[0];
  const float* w1  = (const float*)d_in[1];
  const float* b1  = (const float*)d_in[2];
  const float* w2  = (const float*)d_in[3];
  const float* b2  = (const float*)d_in[4];
  float* out = (float*)d_out;

  // d_ws layout: w1t (E,F,D) bf16 | w2t (E,D,F) bf16  => 2 * 16.78 MB = 33.5 MB
  unsigned short* w1t = (unsigned short*)d_ws;
  unsigned short* w2t = w1t + (size_t)EE * DD * FF;

  // pre-pass: transpose+convert weights once (vs 256 redundant conversions/WG before)
  hipLaunchKernelGGL(transpose_cvt_kernel, dim3(FF / 32, DD / 32, EE), dim3(32, 8),
                     0, stream, w1, w1t, DD, FF);
  hipLaunchKernelGGL(transpose_cvt_kernel, dim3(DD / 32, FF / 32, EE), dim3(32, 8),
                     0, stream, w2, w2t, FF, DD);

  dim3 grid(BB * EE * (CC / ROWS));   // 2048 workgroups
  dim3 block(256);                    // 8 wave32s
  hipLaunchKernelGGL(moe_expert_ffn_kernel, grid, block, 0, stream,
                     xin, w1t, b1, w2t, b2, out);
}
